// QRNNClassifier_38388417692513
// MI455X (gfx1250) — matrix operand order repaired and checked by reading, not compiled, not run on hardware
//
#include <hip/hip_runtime.h>
#include <math.h>

typedef __attribute__((ext_vector_type(16))) _Float16 v16h;
typedef __attribute__((ext_vector_type(8)))  float    v8f;
typedef __attribute__((ext_vector_type(4)))  _Float16 v4h;
typedef __attribute__((ext_vector_type(2)))  _Float16 v2h;

#define T_LEN 512
#define BATCH 128
#define EDIM  300
#define HDIM  512
#define NG    1536               // 3*H
#define LCLS  50
#define MTOT  (T_LEN * BATCH)    // 65536 rows of the gates GEMM
#define KPAD  320                // ceil(300/32)*32
#define NKT   (KPAD / 32)        // 10 K-steps
#define MT    128                // M tile
#define NT    128                // N tile
#define LDA   36                 // padded LDS row stride (halves); 36*2=72B, 8B-aligned rows

// ---------------------------------------------------------------------------
// Kernel 0: one-shot pack of W_gates into per-lane WMMA B-fragment layout.
// Bpack[((nt16*NKT + kt)*32 + lane)*16 + i] = (f16) Wg[kt*32 + kb + i][nt16*16 + lane%16]
//   with kb = (lane<16 ? 0 : 16), zero-padded for k >= 300.
// One contiguous 32B chunk per (frag, lane) -> GEMM B loads are 2x b128, coalesced.
// ---------------------------------------------------------------------------
__global__ __launch_bounds__(256)
void qrnn_pack_Wg(const float* __restrict__ Wg, _Float16* __restrict__ Bpack)
{
    const int t = blockIdx.x * blockDim.x + threadIdx.x;   // (NG/16)*NKT*32 = 30720
    if (t >= (NG / 16) * NKT * 32) return;
    const int lane = t & 31;
    const int kt   = (t >> 5) % NKT;
    const int nt16 = t / (32 * NKT);
    const int n    = nt16 * 16 + (lane & 15);
    const int kb   = kt * 32 + ((lane < 16) ? 0 : 16);
    _Float16* dst = Bpack + (size_t)t * 16;
    #pragma unroll
    for (int i = 0; i < 16; ++i) {
        const int k = kb + i;
        dst[i] = (_Float16)((k < EDIM) ? Wg[(size_t)k * NG + n] : 0.0f);
    }
}

static __device__ __forceinline__ v4h cvt4(float a, float b, float c, float d) {
    v4h r; r[0] = (_Float16)a; r[1] = (_Float16)b; r[2] = (_Float16)c; r[3] = (_Float16)d;
    return r;
}

// ---------------------------------------------------------------------------
// Kernel 1: fused embedding-gather + gates GEMM (f16 WMMA, f32 accum)
//           + bias + activations, streaming z/f (f16) and o at t=T-1 (f32).
// A tile double-buffered in LDS; B fragments straight from L2-resident Bpack.
// ---------------------------------------------------------------------------
__global__ __launch_bounds__(256)
void qrnn_gates_gemm(const int*      __restrict__ sentence,
                     const float*    __restrict__ emb,
                     const _Float16* __restrict__ Bpack,
                     const float*    __restrict__ bg,
                     _Float16*       __restrict__ z_buf,
                     _Float16*       __restrict__ f_buf,
                     float*          __restrict__ o_last)
{
    __shared__ _Float16 A_lds[2][MT * LDA];
    __shared__ int      s_idx[MT];

    const int tid  = threadIdx.x;
    const int lane = tid & 31;
    const int wid  = tid >> 5;     // 0..7 (8 wave32s)
    const int wm   = wid & 3;      // 4 wave rows  -> 32 M rows each
    const int wn   = wid >> 2;     // 2 wave cols  -> 64 N cols each
    const int m0   = blockIdx.x * MT;
    const int n0   = blockIdx.y * NT;

    if (tid < MT) s_idx[tid] = sentence[m0 + tid];
    __syncthreads();

    // per-thread gather assignment: 2 threads per row, 16 contiguous cols each
    const int ar = tid >> 1;
    const int ac = (tid & 1) << 4;
    const float* row0 = emb + (size_t)s_idx[ar] * EDIM;   // loop-invariant: hoisted

    // ---- stage K-step 0 into buffer 0 (cols 0..31, all < 300: guard-free)
    {
        const float4* s = (const float4*)(row0 + ac);
        const float4 a0 = s[0], a1 = s[1], a2 = s[2], a3 = s[3];
        v4h* d = (v4h*)&A_lds[0][ar * LDA + ac];
        d[0] = cvt4(a0.x, a0.y, a0.z, a0.w);
        d[1] = cvt4(a1.x, a1.y, a1.z, a1.w);
        d[2] = cvt4(a2.x, a2.y, a2.z, a2.w);
        d[3] = cvt4(a3.x, a3.y, a3.z, a3.w);
    }
    __syncthreads();

    v8f acc[8] = {};                       // 2 (M) x 4 (N) tiles of 16x16 f32
    const v16h* Bp = (const v16h*)Bpack;
    const int   ntb = (n0 >> 4) + wn * 4;  // base 16-col fragment index for this wave

    int buf = 0;
    for (int kt = 0; kt < NKT; ++kt) {
        // ---- prefetch next A tile into registers (overlaps with WMMA below)
        float nx[16];
        if (kt + 1 < NKT) {
            const int k1 = (kt + 1) * 32;
            if (k1 + ac + 15 < EDIM) {                 // K-steps 1..8: guard-free float4s
                const float4* s = (const float4*)(row0 + k1 + ac);
                const float4 a0 = s[0], a1 = s[1], a2 = s[2], a3 = s[3];
                nx[0]=a0.x; nx[1]=a0.y; nx[2]=a0.z; nx[3]=a0.w;
                nx[4]=a1.x; nx[5]=a1.y; nx[6]=a1.z; nx[7]=a1.w;
                nx[8]=a2.x; nx[9]=a2.y; nx[10]=a2.z; nx[11]=a2.w;
                nx[12]=a3.x; nx[13]=a3.y; nx[14]=a3.z; nx[15]=a3.w;
            } else {                                   // tail: clamp address, VALU select
                #pragma unroll
                for (int c = 0; c < 16; ++c) {
                    const int k  = k1 + ac + c;
                    const int kc = (k < EDIM) ? k : (EDIM - 1);
                    const float v = row0[kc];          // always in-bounds, no exec dance
                    nx[c] = (k < EDIM) ? v : 0.0f;
                }
            }
        }

        // ---- fragments (contiguous k = 16h + s, same order in A and Bpack)
        v16h afrag[2], bfrag[4];
        {
            // A 16x32: lanes 0-15 -> M=lane, K 0..15; lanes 16-31 -> K 16..31
            const int arow = wm * 32 + (lane & 15);
            const int akb  = (lane < 16) ? 0 : 16;
            #pragma unroll
            for (int mi = 0; mi < 2; ++mi) {
                const _Float16* ap = &A_lds[buf][(arow + mi * 16) * LDA + akb];
                #pragma unroll
                for (int i = 0; i < 8; ++i) {
                    afrag[mi][i]     = ap[i];
                    afrag[mi][8 + i] = ap[8 + i];
                }
            }
            #pragma unroll
            for (int nj = 0; nj < 4; ++nj)
                bfrag[nj] = Bp[(size_t)(((ntb + nj) * NKT + kt) * 32 + lane)];
        }

        // ---- 8 WMMAs per K-step per wave
        #pragma unroll
        for (int mi = 0; mi < 2; ++mi)
            #pragma unroll
            for (int nj = 0; nj < 4; ++nj)
                acc[mi * 4 + nj] = __builtin_amdgcn_wmma_f32_16x16x32_f16(
                    false, afrag[mi], false, bfrag[nj],
                    (short)0, acc[mi * 4 + nj], false, false);

        // ---- commit prefetched tile into the other buffer
        if (kt + 1 < NKT) {
            v4h* d = (v4h*)&A_lds[buf ^ 1][ar * LDA + ac];
            d[0] = cvt4(nx[0],  nx[1],  nx[2],  nx[3]);
            d[1] = cvt4(nx[4],  nx[5],  nx[6],  nx[7]);
            d[2] = cvt4(nx[8],  nx[9],  nx[10], nx[11]);
            d[3] = cvt4(nx[12], nx[13], nx[14], nx[15]);
        }
        __syncthreads();
        buf ^= 1;
    }

    // ---- epilogue: bias + activation, scatter per C layout
    // C: VGPR v, lanes 0-15 -> M=v, N=lane; lanes 16-31 -> M=v+8, N=lane-16
    #pragma unroll
    for (int mi = 0; mi < 2; ++mi) {
        #pragma unroll
        for (int nj = 0; nj < 4; ++nj) {
            const v8f c   = acc[mi * 4 + nj];
            const int n   = n0 + wn * 64 + nj * 16 + (lane & 15);
            const int mrb = m0 + wm * 32 + mi * 16 + ((lane < 16) ? 0 : 8);
            const float bias = bg[n];
            #pragma unroll
            for (int v = 0; v < 8; ++v) {
                const int m = mrb + v;
                const float g = c[v] + bias;
                if (n < HDIM) {                        // z = tanh
                    z_buf[(size_t)m * HDIM + n] = (_Float16)tanhf(g);
                } else if (n < 2 * HDIM) {             // f = sigmoid
                    f_buf[(size_t)m * HDIM + (n - HDIM)] =
                        (_Float16)(1.0f / (1.0f + expf(-g)));
                } else if (m >= MTOT - BATCH) {        // o only at t = T-1
                    o_last[(size_t)(m - (MTOT - BATCH)) * HDIM + (n - 2 * HDIM)] =
                        1.0f / (1.0f + expf(-g));
                }
            }
        }
    }
}

// ---------------------------------------------------------------------------
// Kernel 2: fo-pool recurrence, one thread per (b, h-pair); only c_{T-1} kept.
// ---------------------------------------------------------------------------
__global__ __launch_bounds__(256)
void qrnn_fo_pool(const _Float16* __restrict__ z_buf,
                  const _Float16* __restrict__ f_buf,
                  const float*    __restrict__ o_last,
                  float*          __restrict__ h_last)
{
    const int tid = blockIdx.x * blockDim.x + threadIdx.x;   // 0 .. B*H/2-1
    const int b  = tid >> 8;                                 // 256 h-pairs per batch row
    const int hp = (tid & 255) << 1;
    float c0 = 0.0f, c1 = 0.0f;
    for (int t = 0; t < T_LEN; ++t) {
        const int off = ((t * BATCH + b) << 9) + hp;         // 4B-aligned, coalesced
        const v2h f2 = *(const v2h*)(f_buf + off);
        const v2h z2 = *(const v2h*)(z_buf + off);
        const float fa = (float)f2[0], fb = (float)f2[1];
        c0 = fmaf(fa, c0, (1.0f - fa) * (float)z2[0]);
        c1 = fmaf(fb, c1, (1.0f - fb) * (float)z2[1]);
    }
    const int o = b * HDIM + hp;
    h_last[o]     = o_last[o]     * c0;
    h_last[o + 1] = o_last[o + 1] * c1;
}

// ---------------------------------------------------------------------------
// Kernel 3: classifier + log-softmax (tiny: 128x50).
// ---------------------------------------------------------------------------
__global__ __launch_bounds__(64)
void qrnn_classify(const float* __restrict__ h_last,
                   const float* __restrict__ W_out,
                   const float* __restrict__ b_out,
                   float*       __restrict__ out)
{
    __shared__ float hrow[HDIM];
    __shared__ float sy[64];
    __shared__ float red[2];
    const int b = blockIdx.x;
    const int t = threadIdx.x;

    for (int i = t; i < HDIM; i += 64) hrow[i] = h_last[b * HDIM + i];
    __syncthreads();

    float y = -1e30f;
    if (t < LCLS) {
        float acc = b_out[t];
        for (int k = 0; k < HDIM; ++k) acc = fmaf(hrow[k], W_out[k * LCLS + t], acc);
        y = acc;
    }
    sy[t] = y;
    __syncthreads();
    if (t == 0) {
        float mx = -1e30f;
        for (int i = 0; i < LCLS; ++i) mx = fmaxf(mx, sy[i]);
        red[0] = mx;
    }
    __syncthreads();
    const float mx = red[0];
    sy[t] = (t < LCLS) ? expf(y - mx) : 0.0f;
    __syncthreads();
    if (t == 0) {
        float s = 0.0f;
        for (int i = 0; i < LCLS; ++i) s += sy[i];
        red[1] = logf(s);
    }
    __syncthreads();
    if (t < LCLS) out[b * LCLS + t] = (y - mx) - red[1];
}

// ---------------------------------------------------------------------------
extern "C" void kernel_launch(void* const* d_in, const int* in_sizes, int n_in,
                              void* d_out, int out_size, void* d_ws, size_t ws_size,
                              hipStream_t stream)
{
    const int*   sentence = (const int*)  d_in[0];
    const float* emb      = (const float*)d_in[1];
    const float* Wg       = (const float*)d_in[2];
    const float* bg       = (const float*)d_in[3];
    const float* W_out    = (const float*)d_in[4];
    const float* b_out    = (const float*)d_in[5];
    float* out = (float*)d_out;

    char* ws = (char*)d_ws;
    const size_t zf_bytes = (size_t)MTOT * HDIM * sizeof(_Float16);        // 64 MB each
    const size_t oh_bytes = (size_t)BATCH * HDIM * sizeof(float);          // 256 KB each
    _Float16* z_buf  = (_Float16*)(ws);
    _Float16* f_buf  = (_Float16*)(ws + zf_bytes);
    float*    o_last = (float*)   (ws + 2 * zf_bytes);
    float*    h_last = (float*)   (ws + 2 * zf_bytes + oh_bytes);
    _Float16* Bpack  = (_Float16*)(ws + 2 * zf_bytes + 2 * oh_bytes);      // ~0.94 MB

    const int packN = (NG / 16) * NKT * 32;                                // 30720
    qrnn_pack_Wg<<<(packN + 255) / 256, 256, 0, stream>>>(Wg, Bpack);

    qrnn_gates_gemm<<<dim3(MTOT / MT, NG / NT), 256, 0, stream>>>(
        sentence, emb, Bpack, bg, z_buf, f_buf, o_last);

    qrnn_fo_pool<<<(BATCH * HDIM / 2) / 256, 256, 0, stream>>>(
        z_buf, f_buf, o_last, h_last);

    qrnn_classify<<<BATCH, 64, 0, stream>>>(h_last, W_out, b_out, out);
}
